// SimpleMultiHeadAttention_34454227649167
// MI455X (gfx1250) — matrix operand order repaired and checked by reading, not compiled, not run on hardware
//
#include <hip/hip_runtime.h>
#include <hip/hip_bf16.h>
#include <cstdint>

// ---------------------------------------------------------------------------
// MI455X (gfx1250) multi-head attention forward, bf16 WMMA + fp32 accumulate.
// Async global->LDS staging (ASYNCcnt) in the GEMMs, TDM tensor_load_to_lds
// (TENSORcnt) staging in the flash-attention kernel.
// ---------------------------------------------------------------------------

typedef __attribute__((ext_vector_type(16))) __bf16 v16bf;
typedef __attribute__((ext_vector_type(8)))  __bf16 v8bf;
typedef __attribute__((ext_vector_type(8)))  float  v8f;
typedef __attribute__((ext_vector_type(4)))  unsigned int u32x4;
typedef __attribute__((ext_vector_type(8)))  int i32x8;
typedef __attribute__((ext_vector_type(4)))  int i32x4;

static constexpr int SEQ  = 4096;
static constexpr int DIMC = 1024;
static constexpr int NH   = 16;
static constexpr int HDIM = 64;

#if __has_builtin(__builtin_amdgcn_global_load_async_to_lds_b128)
#define HAVE_ASYNC_LDS 1
#else
#define HAVE_ASYNC_LDS 0
#endif
#if __has_builtin(__builtin_amdgcn_tensor_load_to_lds)
#define HAVE_TDM 1
#else
#define HAVE_TDM 0
#endif

// Builtin signature (from the round-2 diagnostic): param0 is a pointer to
// 'int __vector(4)' in the global (AS1 / "__device__") address space.
typedef __attribute__((address_space(1))) i32x4 gas_i32x4;
typedef __attribute__((address_space(3))) i32x4 las_i32x4;

// Per-lane async 16B copy global -> LDS (tracked by ASYNCcnt).
__device__ __forceinline__ void async_cp16(const void* g, void* l) {
#if HAVE_ASYNC_LDS
    __builtin_amdgcn_global_load_async_to_lds_b128(
        (gas_i32x4*)(uintptr_t)g, (las_i32x4*)(uint32_t)(uintptr_t)l, 0, 0);
#else
    *(v8bf*)l = *(const v8bf*)g;
#endif
}

__device__ __forceinline__ void wait_async_le4() {
#if HAVE_ASYNC_LDS
#if __has_builtin(__builtin_amdgcn_s_wait_asynccnt)
    __builtin_amdgcn_s_wait_asynccnt(4);
#else
    asm volatile("s_wait_asynccnt 0x4" ::: "memory");
#endif
#endif
}
__device__ __forceinline__ void wait_async_0() {
#if HAVE_ASYNC_LDS
#if __has_builtin(__builtin_amdgcn_s_wait_asynccnt)
    __builtin_amdgcn_s_wait_asynccnt(0);
#else
    asm volatile("s_wait_asynccnt 0x0" ::: "memory");
#endif
#endif
}

#if HAVE_TDM
// TDM: 2D bf16 tile load, rows tile_d0 elems, tile_d1 rows, global row stride
// stride0 elems; LDS destination padded 4 DWORDs every 32 DWORDs (64 bf16 row
// -> 72 bf16 padded row, matching LDF below).
__device__ __forceinline__ void tdm_load_tile_2d(void* lds_dst, const void* gsrc,
                                                 unsigned tile_d0, unsigned tile_d1,
                                                 unsigned long long stride0) {
    unsigned long long ga = (unsigned long long)(uintptr_t)gsrc;
    u32x4 g0;
    g0.x = 1u;                                   // count=1, user descriptor
    g0.y = (uint32_t)(uintptr_t)lds_dst;         // LDS byte address (low 32 bits)
    g0.z = (uint32_t)ga;                         // global_addr[31:0]
    g0.w = ((uint32_t)(ga >> 32) & 0x01FFFFFFu)  // global_addr[56:32]
           | (2u << 30);                         // type = 2 ("image")
    i32x8 g1;
    g1[0] = (int)((1u << 16)                     // data_size = 1 -> 2 bytes
                | (1u << 20)                     // pad_enable
                | (4u << 22)                     // pad_interval: 32 DWORDs (128B)
                | (3u << 25));                   // pad_amount: 4 DWORDs (16B)
    g1[1] = (int)((tile_d0 & 0xFFFFu) << 16);    // tensor_dim0 = tile_d0
    g1[2] = (int)((tile_d1 & 0xFFFFu) << 16);    // tensor_dim1 = tile_d1
    g1[3] = (int)((tile_d0 & 0xFFFFu) << 16);    // tile_dim0
    g1[4] = (int)(tile_d1 & 0xFFFFu);            // tile_dim1 (tile_dim2 = 0)
    g1[5] = (int)(uint32_t)(stride0 & 0xFFFFFFFFu);        // dim0 stride lo
    g1[6] = (int)(uint32_t)((stride0 >> 32) & 0xFFFFu);    // dim0 stride hi
    g1[7] = 0;
    i32x4 z4 = {0, 0, 0, 0};
#if __clang_major__ >= 23
    i32x8 z8 = {0, 0, 0, 0, 0, 0, 0, 0};
    __builtin_amdgcn_tensor_load_to_lds(g0, g1, z4, z4, z8, 0);
#else
    __builtin_amdgcn_tensor_load_to_lds(g0, g1, z4, z4, 0);
#endif
}
#endif

// Build a 16-element bf16 fragment from two 8-element (16B) chunks.
__device__ __forceinline__ v16bf ld_frag(const __bf16* p0, const __bf16* p1) {
    v8bf a = *(const v8bf*)p0;
    v8bf b = *(const v8bf*)p1;
    return __builtin_shufflevector(a, b, 0,1,2,3,4,5,6,7,8,9,10,11,12,13,14,15);
}

__device__ __forceinline__ v8f wmma_bf16(v16bf a, v16bf b, v8f c) {
    return __builtin_amdgcn_wmma_f32_16x16x32_bf16(
        false, a, false, b, (short)0, c, false, false);
}

// ---------------------------------------------------------------------------
// Elementwise conversion kernels
// ---------------------------------------------------------------------------
__global__ __launch_bounds__(256) void cvt_f32_bf16(const float* __restrict__ src,
                                                    __bf16* __restrict__ dst, int n) {
    int i = blockIdx.x * 256 + threadIdx.x;
    if (i < n) dst[i] = (__bf16)src[i];
}

// W[K][N] (row-major) -> WT[N][K] bf16 (row-major), i.e. N-major for B-fragments.
__global__ __launch_bounds__(256) void transpose_cvt(const float* __restrict__ W,
                                                     __bf16* __restrict__ WT,
                                                     int K, int N) {
    int i = blockIdx.x * 256 + threadIdx.x;
    if (i >= K * N) return;
    int k = i / N;
    int n = i - k * N;
    WT[(size_t)n * K + k] = (__bf16)W[(size_t)k * N + n];
}

// ---------------------------------------------------------------------------
// Generic bf16 GEMM:  C[M][N] (f32) = A[M][K] * BT[N][K]^T
// Block tile 128x128, 8 waves; wave tile 32x64 (2x4 WMMA accumulators).
// Double-buffered LDS, async global->LDS staging overlapped with WMMA.
// ---------------------------------------------------------------------------
static constexpr int LDT = 40;   // padded LDS row stride in bf16 (80B, 16B-aligned)

__global__ __launch_bounds__(256) void gemm_bf16_nt(const __bf16* __restrict__ A,
                                                    const __bf16* __restrict__ BT,
                                                    float* __restrict__ C,
                                                    int M, int N, int K) {
    (void)M;
    __shared__ __attribute__((aligned(16))) __bf16 As[2][128 * LDT];
    __shared__ __attribute__((aligned(16))) __bf16 Bs[2][128 * LDT];

    const int tid  = threadIdx.x;
    const int wave = tid >> 5;
    const int lane = tid & 31;
    const int hi   = lane >> 4;
    const int lo   = lane & 15;

    const int bm = blockIdx.y * 128;
    const int bn = blockIdx.x * 128;
    const int wm = (wave & 3) * 32;
    const int wn = (wave >> 2) * 64;

    v8f acc[2][4];
#pragma unroll
    for (int mt = 0; mt < 2; ++mt)
#pragma unroll
        for (int nt = 0; nt < 4; ++nt) acc[mt][nt] = v8f{};

    const int srow = tid >> 2;        // 0..63
    const int sch  = (tid & 3) * 8;   // bf16 chunk offset

    // Stage one 128x32 K-tile pair into buffer `buf` (4 async 16B copies/lane).
    auto stage = [&](int k0, int buf) {
#pragma unroll
        for (int p = 0; p < 2; ++p) {
            int row = srow + p * 64;
            async_cp16(&A[(size_t)(bm + row) * K + k0 + sch], &As[buf][row * LDT + sch]);
            async_cp16(&BT[(size_t)(bn + row) * K + k0 + sch], &Bs[buf][row * LDT + sch]);
        }
    };

    const int nk = K / 32;
    stage(0, 0);

    for (int i = 0; i < nk; ++i) {
        const int cur = i & 1;
        if (i + 1 < nk) {
            stage((i + 1) * 32, cur ^ 1);      // prefetch next tile into other buffer
            wait_async_le4();                   // current tile's 4 copies complete
        } else {
            wait_async_0();
        }
        __syncthreads();

        v16bf af[2], bf[4];
#pragma unroll
        for (int mt = 0; mt < 2; ++mt) {
            const __bf16* pa = &As[cur][(wm + mt * 16 + lo) * LDT + hi * 8];
            af[mt] = ld_frag(pa, pa + 16);      // K = hi*8..+7 and hi*8+16..+23
        }
#pragma unroll
        for (int nt = 0; nt < 4; ++nt) {
            const __bf16* pb = &Bs[cur][(wn + nt * 16 + lo) * LDT + hi * 8];
            bf[nt] = ld_frag(pb, pb + 16);      // K = hi*8..+7 and hi*8+16..+23
        }
#pragma unroll
        for (int mt = 0; mt < 2; ++mt)
#pragma unroll
            for (int nt = 0; nt < 4; ++nt)
                acc[mt][nt] = wmma_bf16(af[mt], bf[nt], acc[mt][nt]);
        __syncthreads();
    }

    // C layout: M = r + 8*hi, N = lo
#pragma unroll
    for (int mt = 0; mt < 2; ++mt)
#pragma unroll
        for (int nt = 0; nt < 4; ++nt)
#pragma unroll
            for (int r = 0; r < 8; ++r) {
                int row = bm + wm + mt * 16 + 8 * hi + r;
                int col = bn + wn + nt * 16 + lo;
                C[(size_t)row * N + col] = acc[mt][nt][r];
            }
}

// ---------------------------------------------------------------------------
// RoPE + pack: qkv f32 [S][3*DIM] -> Qb/Kb bf16 [H][S][HD], Vt bf16 [H][HD][S]
// ---------------------------------------------------------------------------
__global__ __launch_bounds__(256) void rope_pack(const float* __restrict__ qkv,
                                                 const float* __restrict__ fcos,
                                                 const float* __restrict__ fsin,
                                                 __bf16* __restrict__ Qb,
                                                 __bf16* __restrict__ Kb,
                                                 __bf16* __restrict__ Vt) {
    int i = blockIdx.x * 256 + threadIdx.x;       // (s, h, d2): 4096*16*32
    if (i >= SEQ * NH * (HDIM / 2)) return;
    int d2 = i & 31;
    int h  = (i >> 5) & 15;
    int s  = i >> 9;

    float c  = fcos[s * 32 + d2];
    float sn = fsin[s * 32 + d2];

    size_t base = (size_t)s * (3 * DIMC) + h * HDIM + 2 * d2;
    float qr = qkv[base + 0],          qi = qkv[base + 1];
    float kr = qkv[base + DIMC],       ki = qkv[base + DIMC + 1];
    float vr = qkv[base + 2 * DIMC],   vi = qkv[base + 2 * DIMC + 1];

    float qor = qr * c - qi * sn, qoi = qr * sn + qi * c;
    float kor = kr * c - ki * sn, koi = kr * sn + ki * c;

    size_t qk = ((size_t)h * SEQ + s) * HDIM + 2 * d2;
    Qb[qk + 0] = (__bf16)qor;  Qb[qk + 1] = (__bf16)qoi;
    Kb[qk + 0] = (__bf16)kor;  Kb[qk + 1] = (__bf16)koi;
    Vt[((size_t)h * HDIM + 2 * d2 + 0) * SEQ + s] = (__bf16)vr;
    Vt[((size_t)h * HDIM + 2 * d2 + 1) * SEQ + s] = (__bf16)vi;
}

// ---------------------------------------------------------------------------
// Flash attention: block = (head, 64 query rows); 4 waves x 16 query rows.
// K/V tiles staged by the Tensor Data Mover (wave 0 issues, TENSORcnt wait).
// ---------------------------------------------------------------------------
static constexpr int LDF = 72;    // 144B rows: 16B-aligned, spreads banks

__global__ __launch_bounds__(128) void flash_attn(const __bf16* __restrict__ Qb,
                                                  const __bf16* __restrict__ Kb,
                                                  const __bf16* __restrict__ Vt,
                                                  __bf16* __restrict__ Yb) {
    __shared__ __attribute__((aligned(16))) __bf16 Ks[64 * LDF];
    __shared__ __attribute__((aligned(16))) __bf16 Vs[64 * LDF];
    __shared__ __attribute__((aligned(16))) __bf16 Ps[4 * 16 * LDF];

    const int h  = blockIdx.y;
    const int qb = blockIdx.x;
    const int tid  = threadIdx.x;
    const int wave = tid >> 5;
    const int lane = tid & 31;
    const int hi = lane >> 4;
    const int lo = lane & 15;

    const int qrow = qb * 64 + wave * 16 + lo;
    v16bf qf[2];
#pragma unroll
    for (int kk = 0; kk < 2; ++kk) {
        const __bf16* pq = &Qb[((size_t)h * SEQ + qrow) * HDIM + kk * 32 + hi * 8];
        qf[kk] = ld_frag(pq, pq + 16);
    }

    v8f o[4];
#pragma unroll
    for (int dt = 0; dt < 4; ++dt) o[dt] = v8f{};
    float mrow[8], lrow[8];
#pragma unroll
    for (int r = 0; r < 8; ++r) { mrow[r] = -1e30f; lrow[r] = 0.0f; }

    __bf16* P = &Ps[wave * 16 * LDF];
    const int nkb = qb + 1;           // causal: key blocks 0..qb

    for (int kb = 0; kb < nkb; ++kb) {
        // ---- stage K tile [64 x 64] and Vt tile [64(d) x 64(keys)] ----
#if HAVE_TDM
        if (wave == 0) {
            tdm_load_tile_2d(Ks, &Kb[((size_t)h * SEQ + kb * 64) * HDIM], 64, 64, HDIM);
            tdm_load_tile_2d(Vs, &Vt[(size_t)h * HDIM * SEQ + (size_t)kb * 64], 64, 64, SEQ);
            __builtin_amdgcn_s_wait_tensorcnt(0);
        }
#else
#pragma unroll
        for (int p = 0; p < 4; ++p) {
            int row = (tid >> 3) + p * 16;
            int ch  = (tid & 7) * 8;
            *(v8bf*)&Ks[row * LDF + ch] =
                *(const v8bf*)&Kb[((size_t)h * SEQ + kb * 64 + row) * HDIM + ch];
            *(v8bf*)&Vs[row * LDF + ch] =
                *(const v8bf*)&Vt[((size_t)h * HDIM + row) * SEQ + (size_t)kb * 64 + ch];
        }
#endif
        __syncthreads();

        // ---- scores: 4 tiles of 16x16, K-depth = HD = 64 (2 WMMA each) ----
        v8f st[4];
#pragma unroll
        for (int nt = 0; nt < 4; ++nt) {
            v8f a = v8f{};
#pragma unroll
            for (int kk = 0; kk < 2; ++kk) {
                const __bf16* pk = &Ks[(nt * 16 + lo) * LDF + kk * 32 + hi * 8];
                a = wmma_bf16(qf[kk], ld_frag(pk, pk + 16), a);
            }
            st[nt] = a;
        }

        // ---- scale + causal mask (C layout: M = r + 8*hi, N = lo) ----
        const int qg0 = qb * 64 + wave * 16 + 8 * hi;
#pragma unroll
        for (int nt = 0; nt < 4; ++nt) {
            int kg = kb * 64 + nt * 16 + lo;
#pragma unroll
            for (int r = 0; r < 8; ++r) {
                float sv = st[nt][r] * 0.125f;       // 1/sqrt(64)
                st[nt][r] = (kg > qg0 + r) ? -1e30f : sv;
            }
        }

        // ---- online softmax: row reductions across the 16-lane half ----
        float bmax[8];
#pragma unroll
        for (int r = 0; r < 8; ++r) {
            float m = st[0][r];
#pragma unroll
            for (int nt = 1; nt < 4; ++nt) m = fmaxf(m, st[nt][r]);
            for (int d = 1; d < 16; d <<= 1)
                m = fmaxf(m, __shfl_xor(m, d, 32));
            bmax[r] = m;
        }
        float alpha[8];
#pragma unroll
        for (int r = 0; r < 8; ++r) {
            float mn = fmaxf(mrow[r], bmax[r]);
            alpha[r] = __expf(mrow[r] - mn);
            mrow[r]  = mn;
        }
#pragma unroll
        for (int nt = 0; nt < 4; ++nt)
#pragma unroll
            for (int r = 0; r < 8; ++r)
                st[nt][r] = __expf(st[nt][r] - mrow[r]);
#pragma unroll
        for (int r = 0; r < 8; ++r) {
            float s = st[0][r] + st[1][r] + st[2][r] + st[3][r];
            for (int d = 1; d < 16; d <<= 1)
                s += __shfl_xor(s, d, 32);
            lrow[r] = lrow[r] * alpha[r] + s;
        }
#pragma unroll
        for (int dt = 0; dt < 4; ++dt)
#pragma unroll
            for (int r = 0; r < 8; ++r) o[dt][r] *= alpha[r];

        // ---- P (C-layout) -> wave-private LDS row-major bf16 ----
#pragma unroll
        for (int nt = 0; nt < 4; ++nt)
#pragma unroll
            for (int r = 0; r < 8; ++r)
                P[(8 * hi + r) * LDF + nt * 16 + lo] = (__bf16)st[nt][r];

        // ---- O += P @ V  (reload P as A-fragments, V columns contiguous) ----
#pragma unroll
        for (int kk = 0; kk < 2; ++kk) {
            const __bf16* pp = &P[lo * LDF + kk * 32 + hi * 8];
            v16bf pf = ld_frag(pp, pp + 16);
#pragma unroll
            for (int dt = 0; dt < 4; ++dt) {
                const __bf16* pv = &Vs[(dt * 16 + lo) * LDF + kk * 32 + hi * 8];
                o[dt] = wmma_bf16(pf, ld_frag(pv, pv + 16), o[dt]);
            }
        }
        __syncthreads();
    }

    // ---- normalize + write bf16 y in [s][DIM] layout ----
#pragma unroll
    for (int r = 0; r < 8; ++r) {
        float inv = 1.0f / lrow[r];
        int q = qb * 64 + wave * 16 + 8 * hi + r;
#pragma unroll
        for (int dt = 0; dt < 4; ++dt)
            Yb[(size_t)q * DIMC + h * HDIM + dt * 16 + lo] = (__bf16)(o[dt][r] * inv);
    }
}

// ---------------------------------------------------------------------------
// Launch
// ---------------------------------------------------------------------------
extern "C" void kernel_launch(void* const* d_in, const int* in_sizes, int n_in,
                              void* d_out, int out_size, void* d_ws, size_t ws_size,
                              hipStream_t stream) {
    (void)in_sizes; (void)n_in; (void)out_size; (void)ws_size;
    const float* x     = (const float*)d_in[0];
    const float* fcos  = (const float*)d_in[1];
    const float* fsin  = (const float*)d_in[2];
    const float* wqkv  = (const float*)d_in[3];
    const float* wproj = (const float*)d_in[4];
    float* out = (float*)d_out;

    char* ws = (char*)d_ws;
    size_t off = 0;
    __bf16* xbf    = (__bf16*)(ws + off); off += (size_t)SEQ * DIMC * 2;
    __bf16* wqkvT  = (__bf16*)(ws + off); off += (size_t)3 * DIMC * DIMC * 2;
    __bf16* wprojT = (__bf16*)(ws + off); off += (size_t)DIMC * DIMC * 2;
    float*  qkv    = (float*) (ws + off); off += (size_t)SEQ * 3 * DIMC * 4;
    __bf16* Qb     = (__bf16*)(ws + off); off += (size_t)NH * SEQ * HDIM * 2;
    __bf16* Kb     = (__bf16*)(ws + off); off += (size_t)NH * SEQ * HDIM * 2;
    __bf16* Vt     = (__bf16*)(ws + off); off += (size_t)NH * HDIM * SEQ * 2;
    __bf16* Yb     = (__bf16*)(ws + off); off += (size_t)SEQ * DIMC * 2;

    cvt_f32_bf16<<<(SEQ * DIMC + 255) / 256, 256, 0, stream>>>(x, xbf, SEQ * DIMC);
    transpose_cvt<<<(DIMC * 3 * DIMC + 255) / 256, 256, 0, stream>>>(wqkv, wqkvT, DIMC, 3 * DIMC);
    transpose_cvt<<<(DIMC * DIMC + 255) / 256, 256, 0, stream>>>(wproj, wprojT, DIMC, DIMC);

    gemm_bf16_nt<<<dim3(3 * DIMC / 128, SEQ / 128), 256, 0, stream>>>(
        xbf, wqkvT, qkv, SEQ, 3 * DIMC, DIMC);

    rope_pack<<<(SEQ * NH * (HDIM / 2) + 255) / 256, 256, 0, stream>>>(
        qkv, fcos, fsin, Qb, Kb, Vt);

    flash_attn<<<dim3(SEQ / 64, NH), 128, 0, stream>>>(Qb, Kb, Vt, Yb);

    gemm_bf16_nt<<<dim3(DIMC / 128, SEQ / 128), 256, 0, stream>>>(
        Yb, wprojT, out, SEQ, DIMC, DIMC);
}
